// ClassGAT_56530359550317
// MI455X (gfx1250) — matrix operand order repaired and checked by reading, not compiled, not run on hardware
//
#include <hip/hip_runtime.h>
#include <hip/hip_bf16.h>

// ---------------------------------------------------------------------------
// 2-layer GAT for MI455X (gfx1250, wave32, WMMA).
// GEMMs: split-bf16 (hi/lo) WMMA f32_16x16x32_bf16, ~fp32 accuracy, matrix-pipe rate.
// Attention: L2-resident f32 atomics (agg + xl fit in 192MB L2).
// ---------------------------------------------------------------------------

typedef __attribute__((ext_vector_type(16))) __bf16 bf16x16;
typedef __attribute__((ext_vector_type(8)))  __bf16 bf16x8;
typedef __attribute__((ext_vector_type(8)))  float  f32x8;

union FragU { bf16x16 v; bf16x8 h[2]; };
union BfBits { unsigned short u; __bf16 b; };

__device__ inline __bf16 f2bf(float f) {                 // round-to-nearest-even
    unsigned u = __float_as_uint(f);
    unsigned r = (u + 0x7fffu + ((u >> 16) & 1u)) >> 16;
    BfBits bb; bb.u = (unsigned short)r; return bb.b;
}
__device__ inline float bf2f(__bf16 b) {
    BfBits bb; bb.b = b;
    return __uint_as_float(((unsigned)bb.u) << 16);
}

// ---------------------------------------------------------------------------
// GEMM: C[M,N] = A[M,K] @ B[K,N], fp32 in/out, split-bf16 WMMA.
// Block tile 128x128, K-step 32, 8 waves (2x4 of 16x16 WMMA tiles per wave).
// ---------------------------------------------------------------------------
__global__ __launch_bounds__(256)
void gat_gemm_bf16split(const float* __restrict__ A, const float* __restrict__ B,
                        float* __restrict__ C, int M, int K, int N)
{
    __shared__ __bf16 Ah[128][32];
    __shared__ __bf16 Al[128][32];
    __shared__ __bf16 Bh[128][32];   // stored [n][k] so lane reads are contiguous
    __shared__ __bf16 Bl[128][32];

    const int tid   = threadIdx.x;
    const int bM    = blockIdx.y * 128;
    const int bN    = blockIdx.x * 128;
    const int wave  = tid >> 5;
    const int lane  = tid & 31;
    const int lrow  = lane & 15;     // M for A-frags / N for B-frags / N for C
    const int lhalf = lane >> 4;     // K-group selector
    const int wm    = (wave >> 1) * 32;   // 4 wave rows (M)
    const int wn    = (wave & 1) * 64;    // 2 wave cols (N)

    f32x8 acc[2][4];
#pragma unroll
    for (int i = 0; i < 2; ++i)
#pragma unroll
        for (int j = 0; j < 4; ++j) acc[i][j] = {};

    for (int kt = 0; kt < K; kt += 32) {
        // ---- A tile 128x32 fp32 -> bf16 hi/lo (256 thr x 4 float4) ----
#pragma unroll
        for (int i = 0; i < 4; ++i) {
            int idx = tid + i * 256;         // 0..1023
            int r   = idx >> 3;              // row 0..127
            int c4  = (idx & 7) << 2;        // col 0..28 step 4
            int gr  = bM + r;
            float4 v = make_float4(0.f, 0.f, 0.f, 0.f);
            if (gr < M) v = *(const float4*)(A + (size_t)gr * K + kt + c4);
            float vv[4] = {v.x, v.y, v.z, v.w};
#pragma unroll
            for (int j = 0; j < 4; ++j) {
                __bf16 h = f2bf(vv[j]);
                Ah[r][c4 + j] = h;
                Al[r][c4 + j] = f2bf(vv[j] - bf2f(h));
            }
        }
        // ---- B tile 32x128 fp32, stored transposed [n][k] ----
#pragma unroll
        for (int i = 0; i < 4; ++i) {
            int idx = tid + i * 256;
            int kr  = idx >> 5;              // k 0..31
            int c4  = (idx & 31) << 2;       // n 0..124 step 4
            float4 v = *(const float4*)(B + (size_t)(kt + kr) * N + bN + c4);
            float vv[4] = {v.x, v.y, v.z, v.w};
#pragma unroll
            for (int j = 0; j < 4; ++j) {
                __bf16 h = f2bf(vv[j]);
                Bh[c4 + j][kr] = h;
                Bl[c4 + j][kr] = f2bf(vv[j] - bf2f(h));
            }
        }
        __syncthreads();

        // ---- fragments ----
        FragU ah[2], al[2], bh[4], bl[4];
        const int ka = lhalf * 8;            // A: chunks at ka and ka+16
        const int kb = lhalf * 8;            // B: chunks at kb and kb+16
#pragma unroll
        for (int ti = 0; ti < 2; ++ti) {
            int m = wm + ti * 16 + lrow;
            ah[ti].h[0] = *(const bf16x8*)&Ah[m][ka];
            ah[ti].h[1] = *(const bf16x8*)&Ah[m][ka + 16];
            al[ti].h[0] = *(const bf16x8*)&Al[m][ka];
            al[ti].h[1] = *(const bf16x8*)&Al[m][ka + 16];
        }
#pragma unroll
        for (int tj = 0; tj < 4; ++tj) {
            int n = wn + tj * 16 + lrow;
            bh[tj].h[0] = *(const bf16x8*)&Bh[n][kb];
            bh[tj].h[1] = *(const bf16x8*)&Bh[n][kb + 16];
            bl[tj].h[0] = *(const bf16x8*)&Bl[n][kb];
            bl[tj].h[1] = *(const bf16x8*)&Bl[n][kb + 16];
        }
        // ---- split-bf16 WMMA: hi*hi + hi*lo + lo*hi ----
#pragma unroll
        for (int ti = 0; ti < 2; ++ti)
#pragma unroll
            for (int tj = 0; tj < 4; ++tj) {
                acc[ti][tj] = __builtin_amdgcn_wmma_f32_16x16x32_bf16(
                    false, ah[ti].v, false, bh[tj].v, (short)0, acc[ti][tj], false, false);
                acc[ti][tj] = __builtin_amdgcn_wmma_f32_16x16x32_bf16(
                    false, ah[ti].v, false, bl[tj].v, (short)0, acc[ti][tj], false, false);
                acc[ti][tj] = __builtin_amdgcn_wmma_f32_16x16x32_bf16(
                    false, al[ti].v, false, bh[tj].v, (short)0, acc[ti][tj], false, false);
            }
        __syncthreads();
    }

    // ---- epilogue: C/D layout: lane = N, VGPR r -> M = lhalf*8 + r ----
#pragma unroll
    for (int ti = 0; ti < 2; ++ti)
#pragma unroll
        for (int tj = 0; tj < 4; ++tj) {
            int col = bN + wn + tj * 16 + lrow;
#pragma unroll
            for (int r = 0; r < 8; ++r) {
                int row = bM + wm + ti * 16 + lhalf * 8 + r;
                if (row < M) C[(size_t)row * N + col] = acc[ti][tj][r];
            }
        }
}

// ---------------------------------------------------------------------------
// Per-node attention coefficients: as[n,h] = <xl[n,h,:], a_src[h,:]>, same dst.
// One wave32 per (node, head).
// ---------------------------------------------------------------------------
__global__ __launch_bounds__(256)
void gat_node_alpha(const float* __restrict__ xl, const float* __restrict__ a_src,
                    const float* __restrict__ a_dst, float* __restrict__ as,
                    float* __restrict__ ad, int Nn, int H, int C)
{
    int wid  = (int)((blockIdx.x * (size_t)blockDim.x + threadIdx.x) >> 5);
    int lane = threadIdx.x & 31;
    if (wid >= Nn * H) return;
    int n = wid / H, h = wid - n * H;
    const float* xp = xl + (size_t)n * H * C + (size_t)h * C;
    const float* sp = a_src + (size_t)h * C;
    const float* dp = a_dst + (size_t)h * C;
    float accs = 0.f, accd = 0.f;
    for (int c = lane; c < C; c += 32) {
        float xv = xp[c];
        accs += xv * sp[c];
        accd += xv * dp[c];
    }
#pragma unroll
    for (int off = 16; off; off >>= 1) {    // wave32 reduction
        accs += __shfl_down(accs, off, 32);
        accd += __shfl_down(accd, off, 32);
    }
    if (lane == 0) { as[wid] = accs; ad[wid] = accd; }
}

// ---------------------------------------------------------------------------
// Monotonic float<->uint map so atomicMax(u32) implements float segment-max.
// ---------------------------------------------------------------------------
__device__ inline unsigned f2mono(float f) {
    unsigned u = __float_as_uint(f);
    return (u & 0x80000000u) ? ~u : (u | 0x80000000u);
}
__device__ inline float mono2f(unsigned u) {
    unsigned b = (u & 0x80000000u) ? (u & 0x7fffffffu) : ~u;
    return __uint_as_float(b);
}

__global__ __launch_bounds__(256)
void gat_zero_u32(unsigned* __restrict__ p, size_t count)
{
    size_t i = blockIdx.x * (size_t)blockDim.x + threadIdx.x;
    if (i < count) p[i] = 0u;
}

// edge pass 1: e = leaky_relu(as[src]+ad[dst]); segment max via mono atomicMax
__global__ __launch_bounds__(256)
void gat_edge_max(const int* __restrict__ ei, const float* __restrict__ as,
                  const float* __restrict__ ad, float* __restrict__ eraw,
                  unsigned* __restrict__ mmono, int E, int EE, int H)
{
    int t = (int)(blockIdx.x * (size_t)blockDim.x + threadIdx.x);
    if (t >= EE * H) return;
    int e = t / H, h = t - e * H;
    int src, dst;
    if (e < E) { src = ei[e]; dst = ei[E + e]; }   // edge_index is [2,E] row-major
    else       { src = dst = e - E; }              // self loop
    float v = as[src * H + h] + ad[dst * H + h];
    v = (v >= 0.f) ? v : 0.2f * v;                 // leaky_relu(0.2)
    eraw[t] = v;
    atomicMax(&mmono[dst * H + h], f2mono(v));
}

// edge pass 2: e = exp(e - m[dst]); segment sum
__global__ __launch_bounds__(256)
void gat_edge_expsum(const int* __restrict__ ei, float* __restrict__ eb,
                     const unsigned* __restrict__ mmono, float* __restrict__ s,
                     int E, int EE, int H)
{
    int t = (int)(blockIdx.x * (size_t)blockDim.x + threadIdx.x);
    if (t >= EE * H) return;
    int e = t / H, h = t - e * H;
    int dst = (e < E) ? ei[E + e] : (e - E);
    float m = mono2f(mmono[dst * H + h]);
    float v = __expf(eb[t] - m);
    eb[t] = v;
    atomicAdd(&s[dst * H + h], v);
}

// edge pass 3: agg[dst] += xl[src] * (e / (s[dst]+1e-16)); one block per edge.
__global__ __launch_bounds__(256)
void gat_edge_scatter(const int* __restrict__ ei, const float* __restrict__ xl,
                      const float* __restrict__ eb, const float* __restrict__ s,
                      float* __restrict__ agg, int E, int H, int C)
{
    int e = blockIdx.x;
    int src, dst;
    if (e < E) { src = ei[e]; dst = ei[E + e]; }
    else       { src = dst = e - E; }
    const int HC = H * C;
    for (int c = threadIdx.x; c < HC; c += 256) {
        int h = c / C;
        float coef = eb[(size_t)e * H + h] / (s[(size_t)dst * H + h] + 1e-16f);
        atomicAdd(&agg[(size_t)dst * HC + c], xl[(size_t)src * HC + c] * coef);
    }
}

// ---------------------------------------------------------------------------
// Fused bias + LayerNorm (+ optional ReLU) + residual (+ residual bias).
// One block per node.
// ---------------------------------------------------------------------------
__global__ __launch_bounds__(256)
void gat_post_ln(const float* __restrict__ agg, const float* __restrict__ bias,
                 const float* __restrict__ g, const float* __restrict__ be,
                 const float* __restrict__ res, const float* __restrict__ rbias,
                 float* __restrict__ out, int HC, int do_relu)
{
    __shared__ float red[2][8];
    const int n = blockIdx.x;
    const float* a = agg + (size_t)n * HC;
    float sum = 0.f, sq = 0.f;
    for (int c = threadIdx.x; c < HC; c += 256) {
        float v = a[c] + bias[c];
        sum += v; sq += v * v;
    }
#pragma unroll
    for (int off = 16; off; off >>= 1) {
        sum += __shfl_down(sum, off, 32);
        sq  += __shfl_down(sq,  off, 32);
    }
    int wave = threadIdx.x >> 5, lane = threadIdx.x & 31;
    if (lane == 0) { red[0][wave] = sum; red[1][wave] = sq; }
    __syncthreads();
    if (threadIdx.x == 0) {
        float ts = 0.f, tq = 0.f;
#pragma unroll
        for (int i = 0; i < 8; ++i) { ts += red[0][i]; tq += red[1][i]; }
        red[0][0] = ts; red[1][0] = tq;
    }
    __syncthreads();
    float mu   = red[0][0] / (float)HC;
    float var  = red[1][0] / (float)HC - mu * mu;
    float rstd = rsqrtf(var + 1e-5f);
    for (int c = threadIdx.x; c < HC; c += 256) {
        float v = a[c] + bias[c];
        float y = (v - mu) * rstd * g[c] + be[c];
        if (do_relu) y = fmaxf(y, 0.f);
        out[(size_t)n * HC + c] = y + res[(size_t)n * HC + c] + rbias[c];
    }
}

// ---------------------------------------------------------------------------
// Launcher
// ---------------------------------------------------------------------------
extern "C" void kernel_launch(void* const* d_in, const int* in_sizes, int n_in,
                              void* d_out, int out_size, void* d_ws, size_t ws_size,
                              hipStream_t stream)
{
    (void)in_sizes; (void)n_in; (void)out_size; (void)ws_size;
    const int Nn = 20000, E = 160000, EE = 180000;
    const int IN = 768, H1 = 4, C1 = 256, HC1 = 1024, OUT = 768;

    const float* x   = (const float*)d_in[0];
    const int*   ei  = (const int*)  d_in[1];
    const float* W1  = (const float*)d_in[2];
    const float* a_s1= (const float*)d_in[3];
    const float* a_d1= (const float*)d_in[4];
    const float* b1  = (const float*)d_in[5];
    const float* g1  = (const float*)d_in[6];
    const float* be1 = (const float*)d_in[7];
    const float* Wr1 = (const float*)d_in[8];
    const float* br1 = (const float*)d_in[9];
    const float* W2  = (const float*)d_in[10];
    const float* a_s2= (const float*)d_in[11];
    const float* a_d2= (const float*)d_in[12];
    const float* b2  = (const float*)d_in[13];
    const float* g2  = (const float*)d_in[14];
    const float* be2 = (const float*)d_in[15];
    const float* Wr2 = (const float*)d_in[16];
    const float* br2 = (const float*)d_in[17];
    float* out = (float*)d_out;

    // workspace carve-out (~313 MB of float)
    float* ws = (float*)d_ws;
    size_t o = 0;
    auto carve = [&](size_t cnt) { float* p = ws + o; o += cnt; return p; };
    float*    xl1  = carve((size_t)Nn * HC1);
    float*    res1 = carve((size_t)Nn * HC1);
    float*    agg1 = carve((size_t)Nn * HC1);
    float*    agg2 = carve((size_t)Nn * OUT);
    float*    as1  = carve((size_t)Nn * H1);
    float*    ad1  = carve((size_t)Nn * H1);
    unsigned* m1   = (unsigned*)carve((size_t)Nn * H1);
    float*    s1   = carve((size_t)Nn * H1);
    float*    e1   = carve((size_t)EE * H1);
    float*    as2  = carve(Nn);
    float*    ad2  = carve(Nn);
    unsigned* m2   = (unsigned*)carve(Nn);
    float*    s2   = carve(Nn);
    float*    e2   = carve(EE);
    float* h1   = xl1;    // reuse: xl1 dead after scatter1
    float* xl2  = agg1;   // reuse: agg1 dead after post1
    float* res2 = res1;   // reuse: res1 dead after post1

    const int gM = (Nn + 127) / 128;
    auto zgrid = [](size_t c) { return (unsigned)((c + 255) / 256); };

    // ---------------- Layer 1 ----------------
    gat_gemm_bf16split<<<dim3(HC1 / 128, gM), 256, 0, stream>>>(x, W1,  xl1,  Nn, IN, HC1);
    gat_gemm_bf16split<<<dim3(HC1 / 128, gM), 256, 0, stream>>>(x, Wr1, res1, Nn, IN, HC1);

    gat_zero_u32<<<zgrid((size_t)Nn * HC1), 256, 0, stream>>>((unsigned*)agg1, (size_t)Nn * HC1);
    gat_zero_u32<<<zgrid((size_t)Nn * H1),  256, 0, stream>>>(m1, (size_t)Nn * H1);
    gat_zero_u32<<<zgrid((size_t)Nn * H1),  256, 0, stream>>>((unsigned*)s1, (size_t)Nn * H1);

    gat_node_alpha<<<(Nn * H1 + 7) / 8, 256, 0, stream>>>(xl1, a_s1, a_d1, as1, ad1, Nn, H1, C1);
    gat_edge_max   <<<zgrid((size_t)EE * H1), 256, 0, stream>>>(ei, as1, ad1, e1, m1, E, EE, H1);
    gat_edge_expsum<<<zgrid((size_t)EE * H1), 256, 0, stream>>>(ei, e1, m1, s1, E, EE, H1);
    gat_edge_scatter<<<EE, 256, 0, stream>>>(ei, xl1, e1, s1, agg1, E, H1, C1);
    gat_post_ln<<<Nn, 256, 0, stream>>>(agg1, b1, g1, be1, res1, br1, h1, HC1, 1);

    // ---------------- Layer 2 ----------------
    gat_gemm_bf16split<<<dim3(OUT / 128, gM), 256, 0, stream>>>(h1, W2,  xl2,  Nn, HC1, OUT);
    gat_gemm_bf16split<<<dim3(OUT / 128, gM), 256, 0, stream>>>(h1, Wr2, res2, Nn, HC1, OUT);

    gat_zero_u32<<<zgrid((size_t)Nn * OUT), 256, 0, stream>>>((unsigned*)agg2, (size_t)Nn * OUT);
    gat_zero_u32<<<zgrid((size_t)Nn), 256, 0, stream>>>(m2, (size_t)Nn);
    gat_zero_u32<<<zgrid((size_t)Nn), 256, 0, stream>>>((unsigned*)s2, (size_t)Nn);

    gat_node_alpha<<<(Nn + 7) / 8, 256, 0, stream>>>(xl2, a_s2, a_d2, as2, ad2, Nn, 1, OUT);
    gat_edge_max   <<<zgrid((size_t)EE), 256, 0, stream>>>(ei, as2, ad2, e2, m2, E, EE, 1);
    gat_edge_expsum<<<zgrid((size_t)EE), 256, 0, stream>>>(ei, e2, m2, s2, E, EE, 1);
    gat_edge_scatter<<<EE, 256, 0, stream>>>(ei, xl2, e2, s2, agg2, E, 1, OUT);
    gat_post_ln<<<Nn, 256, 0, stream>>>(agg2, b2, g2, be2, res2, br2, out, OUT, 0);
}
